// PostPatchAdaptiveGraphLearner_78615081386003
// MI455X (gfx1250) — matrix operand order repaired and checked by reading, not compiled, not run on hardware
//
#include <hip/hip_runtime.h>
#include <math.h>

typedef __bf16 bf16;
typedef __attribute__((ext_vector_type(16))) __bf16 v16bf;
typedef __attribute__((ext_vector_type(8)))  __bf16 v8bf;
typedef __attribute__((ext_vector_type(4)))  __bf16 v4bf;
typedef __attribute__((ext_vector_type(8)))  float  v8f;
typedef __attribute__((ext_vector_type(4)))  float  v4f;

#define N_NODES 358
#define NODE_DIM 64
#define EMBED_DIM 96
#define G_HEADS 4
#define TOPK 10
#define BATCH 8
#define PATCHES 72
#define HD 24   // head dim = 96/4

// ---------------- reduction helpers ----------------

__device__ inline float wave_max(float v) {
#pragma unroll
  for (int o = 16; o; o >>= 1) v = fmaxf(v, __shfl_xor(v, o, 32));
  return v;
}
__device__ inline float wave_sum(float v) {
#pragma unroll
  for (int o = 16; o; o >>= 1) v += __shfl_xor(v, o, 32);
  return v;
}

__device__ inline float block_reduce_sum(float v, float* buf) {
  int tid = threadIdx.x;
  v = wave_sum(v);
  int w = tid >> 5, nw = (blockDim.x + 31) >> 5;
  if ((tid & 31) == 0) buf[w] = v;
  __syncthreads();
  if (tid < 32) {
    float x = (tid < nw) ? buf[tid] : 0.f;
    x = wave_sum(x);
    if (tid == 0) buf[0] = x;
  }
  __syncthreads();
  float r = buf[0];
  __syncthreads();
  return r;
}

__device__ inline float block_reduce_max(float v, float* buf) {
  int tid = threadIdx.x;
  v = wave_max(v);
  int w = tid >> 5, nw = (blockDim.x + 31) >> 5;
  if ((tid & 31) == 0) buf[w] = v;
  __syncthreads();
  if (tid < 32) {
    float x = (tid < nw) ? buf[tid] : -1e30f;
    x = wave_max(x);
    if (tid == 0) buf[0] = x;
  }
  __syncthreads();
  float r = buf[0];
  __syncthreads();
  return r;
}

// argmax with min-index tiebreak; result left in bufv[0]/bufi[0]
__device__ inline void block_argmax(float v, int i, float* bufv, int* bufi) {
  int tid = threadIdx.x;
#pragma unroll
  for (int o = 16; o; o >>= 1) {
    float ov = __shfl_xor(v, o, 32);
    int   oi = __shfl_xor(i, o, 32);
    if (ov > v || (ov == v && oi < i)) { v = ov; i = oi; }
  }
  int w = tid >> 5, nw = (blockDim.x + 31) >> 5;
  if ((tid & 31) == 0) { bufv[w] = v; bufi[w] = i; }
  __syncthreads();
  if (tid == 0) {
    float bv = bufv[0]; int bi = bufi[0];
    for (int k = 1; k < nw; ++k)
      if (bufv[k] > bv || (bufv[k] == bv && bufi[k] < bi)) { bv = bufv[k]; bi = bufi[k]; }
    bufv[0] = bv; bufi[0] = bi;
  }
  __syncthreads();
}

__device__ inline float gelu_exact(float x) {
  return 0.5f * x * (1.f + erff(x * 0.70710678118654752440f));
}
__device__ inline float sigmoidf_(float x) { return 1.f / (1.f + __expf(-x)); }

// row[] holds softmax probs (len n). Select top-k, renormalize, write to out.
__device__ inline void topk_renorm_write(float* row, float* mark, int n,
                                         float* redv, int* redi, float* ssum,
                                         float* out) {
  int tid = threadIdx.x, bs = blockDim.x;
  for (int j = tid; j < n; j += bs) mark[j] = 0.f;
  if (tid == 0) *ssum = 0.f;
  __syncthreads();
  for (int it = 0; it < TOPK; ++it) {
    float bv = -1e30f; int bi = 0x7fffffff;
    for (int j = tid; j < n; j += bs) {
      if (mark[j] == 0.f) {
        float v = row[j];
        if (v > bv || (v == bv && j < bi)) { bv = v; bi = j; }
      }
    }
    block_argmax(bv, bi, redv, redi);
    if (tid == 0) { int a = redi[0]; mark[a] = 1.f; *ssum += row[a]; }
    __syncthreads();
  }
  float inv = 1.f / (*ssum + 1e-8f);
  for (int j = tid; j < n; j += bs) out[j] = mark[j] * row[j] * inv;
}

// A-fragment (16x32 bf16) per-lane load: two contiguous 8-element chunks.
__device__ inline v16bf ldsA(const bf16* p) {
  v8bf lo = *(const v8bf*)p;
  v8bf hi = *(const v8bf*)(p + 16);
  return __builtin_shufflevector(lo, hi, 0, 1, 2, 3, 4, 5, 6, 7,
                                 8, 9, 10, 11, 12, 13, 14, 15);
}

// ---------------- kernel 0: weight pre-packing -------------------------------
// WqkvT_pad: bf16 [mat][head][n=32][k=96], n>=24 zeroed (B operand, lane-contiguous)
// bq_pad:    f32  [mat][head][32], n>=24 zeroed
// wom:       f32  [head*32+j] = mean_d Wo[d][h*24+j]  (j<24, else 0)
// bom:       f32  mean(bo)

__global__ __launch_bounds__(256) void prep_kernel(
    const float* __restrict__ Wqkv, const float* __restrict__ bqkv,
    const float* __restrict__ Wo, const float* __restrict__ bo,
    bf16* __restrict__ wq_t, float* __restrict__ bq_pad,
    float* __restrict__ wom, float* __restrict__ bom) {
  int idx = blockIdx.x * 256 + threadIdx.x;
  if (idx < 3 * 4 * 32 * 96) {
    int k = idx % 96; int t = idx / 96;
    int n = t % 32; t /= 32;
    int h = t % 4; int mat = t / 4;
    float v = (n < 24) ? Wqkv[(size_t)(mat * 96 + h * 24 + n) * 96 + k] : 0.f;
    wq_t[idx] = (bf16)v;
  }
  if (idx < 384) {
    int n = idx % 32; int t = idx / 32;
    int h = t % 4; int mat = t / 4;
    bq_pad[idx] = (n < 24) ? bqkv[mat * 96 + h * 24 + n] : 0.f;
  }
  if (idx < 128) {
    int j = idx % 32, h = idx / 32;
    float s = 0.f;
    if (j < 24)
      for (int d = 0; d < 96; ++d) s += Wo[(size_t)d * 96 + h * 24 + j];
    wom[idx] = s * (1.0f / 96.0f);
  }
  if (idx == 0) {
    float s = 0.f;
    for (int d = 0; d < 96; ++d) s += bo[d];
    bom[0] = s * (1.0f / 96.0f);
  }
}

// ---------------- kernel 1: static adjacencies (relu-matmul, softmax, topk) ----

__global__ __launch_bounds__(128) void static_kernel(
    const float* __restrict__ le1, const float* __restrict__ le2,
    const float* __restrict__ ge1, const float* __restrict__ ge2,
    const float* __restrict__ temp, float* __restrict__ out_static) {
  int h = blockIdx.x / N_NODES, i = blockIdx.x % N_NODES;
  __shared__ float e1[64];
  __shared__ float row[N_NODES];
  __shared__ float mark[N_NODES];
  __shared__ float redv[8];
  __shared__ int   redi[8];
  __shared__ float ssum;
  int tid = threadIdx.x;

  int K; const float* E1; const float* E2; float t;
  if (h < 2) {
    K = 32; E1 = le1 + ((size_t)h * N_NODES + i) * 32; E2 = le2 + (size_t)h * 32 * N_NODES;
    t = fminf(fmaxf(temp[h] * 2.0f, 0.1f), 5.0f);
  } else {
    int g = h - 2;
    K = 64; E1 = ge1 + ((size_t)g * N_NODES + i) * 64; E2 = ge2 + (size_t)g * 64 * N_NODES;
    t = fminf(fmaxf(temp[h] * 0.5f, 0.1f), 2.0f);
  }
  if (tid < K) e1[tid] = E1[tid];
  __syncthreads();
  float invt = 1.f / t;
  for (int j = tid; j < N_NODES; j += 128) {
    float s = 0.f;
    for (int k = 0; k < K; ++k) s += e1[k] * E2[k * N_NODES + j];
    row[j] = fmaxf(s, 0.f) * invt;
  }
  __syncthreads();
  float mx = -1e30f;
  for (int j = tid; j < N_NODES; j += 128) mx = fmaxf(mx, row[j]);
  mx = block_reduce_max(mx, redv);
  float sm = 0.f;
  for (int j = tid; j < N_NODES; j += 128) { float e = __expf(row[j] - mx); row[j] = e; sm += e; }
  sm = block_reduce_sum(sm, redv);
  float inv = 1.f / sm;
  for (int j = tid; j < N_NODES; j += 128) row[j] *= inv;
  __syncthreads();
  topk_renorm_write(row, mark, N_NODES, redv, redi, &ssum,
                    out_static + ((size_t)h * N_NODES + i) * N_NODES);
}

// ---------------- kernel 2: MHA per node (bf16 WMMA, branch-free frags) -------
// One block per (b,n). 256 threads = 8 waves.

__global__ __launch_bounds__(256) void mha_kernel(
    const float* __restrict__ patch, const float* __restrict__ pos_enc,
    const bf16* __restrict__ wq_t, const float* __restrict__ bq_pad,
    const float* __restrict__ womg, const float* __restrict__ bomg,
    const float* __restrict__ Wo, const float* __restrict__ bo,
    const float* __restrict__ tn_g, const float* __restrict__ tn_b,
    float* __restrict__ node_out, float* __restrict__ pmean_out) {
  __shared__ __align__(32) bf16 Xs[80][96];    // x = pr+pos (rows 72..79 = 0)
  __shared__ __align__(32) bf16 Qs[80][32];    // cols 24..31 = 0 (padded bias)
  __shared__ __align__(32) bf16 Ksm[80][32];   // cols 24..31 = 0
  __shared__ __align__(32) bf16 VsmT[32][96];  // [channel][patch]; rows 24..31=0, cols 80..95=0
  __shared__ __align__(16) float SsT[80][16];  // scores transposed [key j][row]
  __shared__ __align__(32) bf16 Sb[16][96];    // softmax probs, cols 72..95 = 0
  __shared__ __align__(16) bf16 AttT[128][80]; // [h*32+channel][patch]
  __shared__ float pmean_s[96];
  __shared__ float rowm[80];
  __shared__ float nodev[96];
  __shared__ float zS[128];
  __shared__ float womS[128];
  __shared__ float red[8];

  int bn = blockIdx.x;
  const float* pf = patch + (size_t)bn * 72 * 96;
  int tid = threadIdx.x, lane = tid & 31, wv = tid >> 5;

  // ---- stage 0: fill X (vectorized), pmean, zero pad regions ----
  for (int idx = tid; idx < 72 * 24; idx += 256) {
    int p = idx / 24, d4 = (idx % 24) * 4;
    v4f a = *(const v4f*)(pf + p * 96 + d4);
    v4f e = *(const v4f*)(pos_enc + p * 96 + d4);
    v4bf o;
    o[0] = (bf16)(a[0] + e[0]); o[1] = (bf16)(a[1] + e[1]);
    o[2] = (bf16)(a[2] + e[2]); o[3] = (bf16)(a[3] + e[3]);
    *(v4bf*)&Xs[p][d4] = o;
  }
  if (tid < 96) {
    float s = 0.f;
    for (int p = 0; p < 72; ++p) s += pf[p * 96 + tid];
    float pm = s * (1.0f / 72.0f);
    pmean_s[tid] = pm;
    pmean_out[(size_t)bn * 96 + tid] = pm;
  }
  if (tid < 128) womS[tid] = womg[tid];
  // zero pads: Xs rows 72..79 (768), VsmT cols 80..95 (512), Sb cols 72..95 (384)
  for (int idx = tid; idx < 768; idx += 256) Xs[72 + idx / 96][idx % 96] = (bf16)0.f;
  for (int idx = tid; idx < 512; idx += 256) VsmT[idx / 16][80 + (idx % 16)] = (bf16)0.f;
  for (int idx = tid; idx < 384; idx += 256) Sb[idx / 24][72 + (idx % 24)] = (bf16)0.f;
  float bom = bomg[0];
  __syncthreads();

  for (int h = 0; h < 4; ++h) {
    // --- QKV: X(72x96) @ W^T(96x24->32pad); 3 mats * 5 Mtiles * 2 Ntiles ---
    for (int t = wv; t < 30; t += 8) {
      int mat = t / 10;
      int mt = (t % 10) / 2;
      int nt = t % 2;
      int m0 = mt * 16, n0 = nt * 16;
      int row = m0 + (lane & 15);
      int koff = (lane >> 4) * 8;
      int col = n0 + (lane & 15);
      int k2l = (lane >> 4) * 8;
      const bf16* arow = &Xs[row][0];
      const bf16* brow = wq_t + ((size_t)((mat * 4 + h) * 32 + col)) * 96;
      v8f acc = {0.f, 0.f, 0.f, 0.f, 0.f, 0.f, 0.f, 0.f};
#pragma unroll
      for (int kk = 0; kk < 96; kk += 32) {
        v16bf a = ldsA(arow + kk + koff);
        v16bf b = ldsA(brow + kk + k2l);
        acc = __builtin_amdgcn_wmma_f32_16x16x32_bf16(false, a, false, b, (short)0, acc, false, false);
      }
      int mb = m0 + (lane >> 4) * 8;
      float bias = bq_pad[(mat * 4 + h) * 32 + col];
      if (mat == 0) {
#pragma unroll
        for (int r = 0; r < 8; ++r) Qs[mb + r][col] = (bf16)(acc[r] + bias);
      } else if (mat == 1) {
#pragma unroll
        for (int r = 0; r < 8; ++r) Ksm[mb + r][col] = (bf16)(acc[r] + bias);
      } else {
        v8bf vv;
#pragma unroll
        for (int r = 0; r < 8; ++r) vv[r] = (bf16)(acc[r] + bias);
        *(v8bf*)&VsmT[col][mb] = vv;  // transposed: [channel][patch]
      }
    }
    __syncthreads();

    // --- attention per 16-row tile ---
    for (int rt = 0; rt < 5; ++rt) {
      int m0 = rt * 16;
      // scores = Q_tile(16x32) @ K^T(32x80); 5 col tiles among waves 0..4
      if (wv < 5) {
        int n0 = wv * 16;
        int row = m0 + (lane & 15);
        int koff = (lane >> 4) * 8;
        int jcol = n0 + (lane & 15);
        int k2l = (lane >> 4) * 8;
        v16bf a = ldsA(&Qs[row][koff]);
        v16bf b = ldsA(&Ksm[jcol][k2l]);
        v8f acc = {0.f, 0.f, 0.f, 0.f, 0.f, 0.f, 0.f, 0.f};
        acc = __builtin_amdgcn_wmma_f32_16x16x32_bf16(false, a, false, b, (short)0, acc, false, false);
        const float scale = 0.20412414523193152f;  // 1/sqrt(24)
        int mb = (lane >> 4) * 8;
        v4f lo = {acc[0] * scale, acc[1] * scale, acc[2] * scale, acc[3] * scale};
        v4f hi = {acc[4] * scale, acc[5] * scale, acc[6] * scale, acc[7] * scale};
        v4f* p = (v4f*)&SsT[jcol][mb];
        p[0] = lo; p[1] = hi;
      }
      __syncthreads();
      // softmax rows (one wave per 2 rows); SsT is [j][row]
      for (int rr = wv; rr < 16; rr += 8) {
        float mx = -1e30f;
        for (int j = lane; j < 72; j += 32) mx = fmaxf(mx, SsT[j][rr]);
        mx = wave_max(mx);
        float sm = 0.f;
        for (int j = lane; j < 72; j += 32) {
          float e = __expf(SsT[j][rr] - mx); SsT[j][rr] = e; sm += e;
        }
        sm = wave_sum(sm);
        float inv = 1.f / sm;
        for (int j = lane; j < 72; j += 32) Sb[rr][j] = (bf16)(SsT[j][rr] * inv);
      }
      __syncthreads();
      // atth = S(16x96pad) @ V(96x32pad); 2 col tiles among waves 0..1
      if (wv < 2) {
        int n0 = wv * 16;
        int row = lane & 15;
        int koff = (lane >> 4) * 8;
        int col = n0 + (lane & 15);
        int k2l = (lane >> 4) * 8;
        v8f acc = {0.f, 0.f, 0.f, 0.f, 0.f, 0.f, 0.f, 0.f};
#pragma unroll
        for (int kk = 0; kk < 96; kk += 32) {
          v16bf a = ldsA(&Sb[row][kk + koff]);
          v16bf b = ldsA(&VsmT[col][kk + k2l]);
          acc = __builtin_amdgcn_wmma_f32_16x16x32_bf16(false, a, false, b, (short)0, acc, false, false);
        }
        int mb = (lane >> 4) * 8;
        v8bf vv;
#pragma unroll
        for (int r = 0; r < 8; ++r) vv[r] = (bf16)acc[r];
        *(v8bf*)&AttT[h * 32 + col][m0 + mb] = vv;
      }
      __syncthreads();
    }  // rt
  }    // heads

  // --- epilogue ---
  // rowmean[p] = Att[p,:] . mean_d(Wo) + mean(bo)
  for (int p = wv; p < 72; p += 8) {
    float s = 0.f;
    for (int c = lane; c < 128; c += 32) s += (float)AttT[c][p] * womS[c];
    s = wave_sum(s);
    if (lane == 0) rowm[p] = s + bom;
  }
  __syncthreads();
  // imp = softmax over patches
  if (wv == 0) {
    float mx = -1e30f;
    for (int p = lane; p < 72; p += 32) mx = fmaxf(mx, rowm[p]);
    mx = wave_max(mx);
    float sm = 0.f;
    for (int p = lane; p < 72; p += 32) { float e = __expf(rowm[p] - mx); rowm[p] = e; sm += e; }
    sm = wave_sum(sm);
    float inv = 1.f / sm;
    for (int p = lane; p < 72; p += 32) rowm[p] *= inv;
  }
  __syncthreads();
  // z[c] = sum_p imp[p] * Att[p][c]
  if (tid < 128) {
    float s = 0.f;
    for (int p = 0; p < 72; ++p) s += rowm[p] * (float)AttT[tid][p];
    zS[tid] = s;
  }
  __syncthreads();
  // node[d] = z . Wo[d,:] + bo[d] + pmean[d]
  if (tid < 96) {
    float s = bo[tid] + pmean_s[tid];
    const float* wod = Wo + (size_t)tid * 96;
#pragma unroll
    for (int h = 0; h < 4; ++h)
      for (int j = 0; j < 24; ++j) s += zS[h * 32 + j] * wod[h * 24 + j];
    nodev[tid] = s;
  }
  __syncthreads();
  if (wv == 0) {
    float s = 0.f;
    for (int d = lane; d < 96; d += 32) s += nodev[d];
    s = wave_sum(s);
    float mu = s * (1.0f / 96.0f);
    float vv = 0.f;
    for (int d = lane; d < 96; d += 32) { float dd = nodev[d] - mu; vv += dd * dd; }
    vv = wave_sum(vv);
    if (lane == 0) { red[0] = mu; red[1] = rsqrtf(vv * (1.0f / 96.0f) + 1e-5f); }
  }
  __syncthreads();
  if (tid < 96)
    node_out[(size_t)bn * 96 + tid] = (nodev[tid] - red[0]) * red[1] * tn_g[tid] + tn_b[tid];
}

// ---------------- kernel 3: dynamic encoder (Linear-LN-GELU-Linear-LN) -------

__global__ __launch_bounds__(128) void dyn_enc_kernel(
    const float* __restrict__ node, const float* __restrict__ w1, const float* __restrict__ b1,
    const float* __restrict__ g1, const float* __restrict__ bb1,
    const float* __restrict__ w2, const float* __restrict__ b2,
    const float* __restrict__ g2, const float* __restrict__ bb2,
    float* __restrict__ de0) {
  __shared__ float xin[96];
  __shared__ float h1[128];
  __shared__ float red[8];
  int r = blockIdx.x, tid = threadIdx.x;
  if (tid < 96) xin[tid] = node[(size_t)r * 96 + tid];
  __syncthreads();
  float v = b1[tid];
  for (int k = 0; k < 96; ++k) v += xin[k] * w1[(size_t)tid * 96 + k];
  float mu = block_reduce_sum(v, red) * (1.0f / 128.0f);
  float d = v - mu;
  float var = block_reduce_sum(d * d, red) * (1.0f / 128.0f);
  h1[tid] = gelu_exact(d * rsqrtf(var + 1e-5f) * g1[tid] + bb1[tid]);
  __syncthreads();
  bool act = tid < 64;
  float v2 = 0.f;
  if (act) {
    v2 = b2[tid];
    for (int k = 0; k < 128; ++k) v2 += h1[k] * w2[(size_t)tid * 128 + k];
  }
  float mu2 = block_reduce_sum(act ? v2 : 0.f, red) * (1.0f / 64.0f);
  float d2 = act ? (v2 - mu2) : 0.f;
  float var2 = block_reduce_sum(d2 * d2, red) * (1.0f / 64.0f);
  if (act) de0[(size_t)r * 64 + tid] = d2 * rsqrtf(var2 + 1e-5f) * g2[tid] + bb2[tid];
}

// ---------------- kernel 4/5: GNN refinement layer ----------------------------

__global__ __launch_bounds__(256) void gnn_kernel(
    const float* __restrict__ dein, const float* __restrict__ W, const float* __restrict__ bias,
    const float* __restrict__ g, const float* __restrict__ bt, float* __restrict__ deout) {
  int id = blockIdx.x;
  int n = id % N_NODES, b = id / N_NODES;
  const float* deB = dein + (size_t)b * N_NODES * 64;
  __shared__ float x[64];
  __shared__ float row[N_NODES];
  __shared__ float agg[64];
  __shared__ float red[8];
  int tid = threadIdx.x;
  if (tid < 64) x[tid] = deB[(size_t)n * 64 + tid];
  __syncthreads();
  for (int m = tid; m < N_NODES; m += 256) {
    float s = 0.f;
    const float* dm = deB + (size_t)m * 64;
    for (int k = 0; k < 64; ++k) s += x[k] * dm[k];
    row[m] = s * 5.0f;  // /0.2
  }
  __syncthreads();
  float mx = -1e30f;
  for (int m = tid; m < N_NODES; m += 256) mx = fmaxf(mx, row[m]);
  mx = block_reduce_max(mx, red);
  float sm = 0.f;
  for (int m = tid; m < N_NODES; m += 256) { float e = __expf(row[m] - mx); row[m] = e; sm += e; }
  sm = block_reduce_sum(sm, red);
  float inv = 1.f / sm;
  for (int m = tid; m < N_NODES; m += 256) row[m] *= inv;
  __syncthreads();
  if (tid < 64) {
    float s = 0.f;
    for (int m = 0; m < N_NODES; ++m) s += row[m] * deB[(size_t)m * 64 + tid];
    agg[tid] = s;
  }
  __syncthreads();
  bool act = tid < 64;
  float v = 0.f;
  if (act) {
    v = bias[tid];
    for (int k = 0; k < 64; ++k) v += agg[k] * W[(size_t)tid * 64 + k];
  }
  float mu = block_reduce_sum(act ? v : 0.f, red) * (1.0f / 64.0f);
  float d = act ? (v - mu) : 0.f;
  float var = block_reduce_sum(d * d, red) * (1.0f / 64.0f);
  if (act)
    deout[(size_t)b * N_NODES * 64 + (size_t)n * 64 + tid] =
        gelu_exact(d * rsqrtf(var + 1e-5f) * g[tid] + bt[tid]) + x[tid];
}

// ---------------- kernel 6: dynamic adjacency (relu, softmax, topk) -----------

__global__ __launch_bounds__(256) void dyn_adj_kernel(
    const float* __restrict__ de, const float* __restrict__ se2w,
    const float* __restrict__ temp, float* __restrict__ out_dyn) {
  int id = blockIdx.x;           // (b*4 + h)*358 + n
  int n = id % N_NODES;
  int bh = id / N_NODES;
  int h = bh % 4, b = bh / 4;
  __shared__ float x[64];
  __shared__ float row[N_NODES];
  __shared__ float mark[N_NODES];
  __shared__ float redv[8];
  __shared__ int   redi[8];
  __shared__ float ssum;
  int tid = threadIdx.x;
  if (tid < 64) x[tid] = de[((size_t)b * N_NODES + n) * 64 + tid];
  float t = fminf(fmaxf(temp[h], 0.1f), 2.0f);
  float invt = 1.f / t;
  __syncthreads();
  for (int m = tid; m < N_NODES; m += 256) {
    float s = 0.f;
    const float* w = se2w + (size_t)h * 64 * N_NODES + m;
    for (int k = 0; k < 64; ++k) s += x[k] * w[(size_t)k * N_NODES];
    row[m] = fmaxf(s, 0.f) * invt;
  }
  __syncthreads();
  float mx = -1e30f;
  for (int m = tid; m < N_NODES; m += 256) mx = fmaxf(mx, row[m]);
  mx = block_reduce_max(mx, redv);
  float sm = 0.f;
  for (int m = tid; m < N_NODES; m += 256) { float e = __expf(row[m] - mx); row[m] = e; sm += e; }
  sm = block_reduce_sum(sm, redv);
  float inv = 1.f / sm;
  for (int m = tid; m < N_NODES; m += 256) row[m] *= inv;
  __syncthreads();
  topk_renorm_write(row, mark, N_NODES, redv, redi, &ssum,
                    out_dyn + (size_t)id * N_NODES);
}

// ---------------- kernel 7: fusion gate fw = sigmoid(pmean @ gf_w^T + b) ------

__global__ __launch_bounds__(256) void fw_kernel(
    const float* __restrict__ pmean, const float* __restrict__ gf_w,
    const float* __restrict__ gf_b, float* __restrict__ fwb) {
  int idx = blockIdx.x * blockDim.x + threadIdx.x;
  if (idx >= BATCH * N_NODES * 4) return;
  int h = idx & 3, bn = idx >> 2;
  float s = gf_b[h];
  const float* pm = pmean + (size_t)bn * 96;
  const float* w = gf_w + (size_t)h * 96;
  for (int d = 0; d < 96; ++d) s += pm[d] * w[d];
  fwb[idx] = sigmoidf_(s);
}

// ---------------- kernel 8: fusion + edge encoder MLP -------------------------

__global__ __launch_bounds__(256) void fuse_edge_kernel(
    const float* __restrict__ st, const float* __restrict__ dy, const float* __restrict__ fwb,
    const float* __restrict__ w1, const float* __restrict__ b1,
    const float* __restrict__ lg, const float* __restrict__ lb,
    const float* __restrict__ w2, const float* __restrict__ b2,
    const float* __restrict__ w3, const float* __restrict__ b3,
    float* __restrict__ outf) {
  long idx = (long)blockIdx.x * blockDim.x + threadIdx.x;
  if (idx >= (long)BATCH * N_NODES * N_NODES) return;
  int j = (int)(idx % N_NODES);
  long t2 = idx / N_NODES;
  int i = (int)(t2 % N_NODES);
  int b = (int)(t2 / N_NODES);

  float fu[4]; float mean4 = 0.f;
#pragma unroll
  for (int h = 0; h < 4; ++h) {
    float w = fwb[((size_t)b * N_NODES + i) * 4 + h];
    float s = st[((size_t)h * N_NODES + i) * N_NODES + j];
    float d = dy[(((size_t)(b * 4 + h)) * N_NODES + i) * N_NODES + j];
    float f = (1.f - w) * s + w * d;
    fu[h] = f; mean4 += f;
  }
  mean4 *= 0.25f;

  float h1[16]; float m = 0.f;
#pragma unroll
  for (int o = 0; o < 16; ++o) {
    float s = b1[o];
#pragma unroll
    for (int h = 0; h < 4; ++h) s += fu[h] * w1[o * 4 + h];
    h1[o] = s; m += s;
  }
  m *= (1.0f / 16.0f);
  float var = 0.f;
#pragma unroll
  for (int o = 0; o < 16; ++o) { float d = h1[o] - m; var += d * d; }
  var *= (1.0f / 16.0f);
  float inv = rsqrtf(var + 1e-5f);
#pragma unroll
  for (int o = 0; o < 16; ++o) h1[o] = gelu_exact((h1[o] - m) * inv * lg[o] + lb[o]);

  float h2[8];
#pragma unroll
  for (int o = 0; o < 8; ++o) {
    float s = b2[o];
#pragma unroll
    for (int k = 0; k < 16; ++k) s += h1[k] * w2[o * 16 + k];
    h2[o] = gelu_exact(s);
  }
  float ew = b3[0];
#pragma unroll
  for (int k = 0; k < 8; ++k) ew += h2[k] * w3[k];
  outf[idx] = sigmoidf_(ew) * mean4;
}

// ---------------- host launcher ----------------------------------------------

extern "C" void kernel_launch(void* const* d_in, const int* in_sizes, int n_in,
                              void* d_out, int out_size, void* d_ws, size_t ws_size,
                              hipStream_t stream) {
  (void)in_sizes; (void)n_in; (void)out_size; (void)ws_size;
  const float* patch    = (const float*)d_in[0];
  const float* se2      = (const float*)d_in[1];
  const float* le1      = (const float*)d_in[2];
  const float* le2      = (const float*)d_in[3];
  const float* ge1      = (const float*)d_in[4];
  const float* ge2      = (const float*)d_in[5];
  const float* temp     = (const float*)d_in[6];
  const float* in_w     = (const float*)d_in[7];
  const float* in_b     = (const float*)d_in[8];
  const float* out_w    = (const float*)d_in[9];
  const float* out_b    = (const float*)d_in[10];
  const float* tn_g     = (const float*)d_in[11];
  const float* tn_b     = (const float*)d_in[12];
  const float* de_w1    = (const float*)d_in[13];
  const float* de_b1    = (const float*)d_in[14];
  const float* de_ln1_g = (const float*)d_in[15];
  const float* de_ln1_b = (const float*)d_in[16];
  const float* de_w2    = (const float*)d_in[17];
  const float* de_b2    = (const float*)d_in[18];
  const float* de_ln2_g = (const float*)d_in[19];
  const float* de_ln2_b = (const float*)d_in[20];
  const float* pos_enc  = (const float*)d_in[21];
  const float* gnn1_w   = (const float*)d_in[22];
  const float* gnn1_b   = (const float*)d_in[23];
  const float* gnn1_g   = (const float*)d_in[24];
  const float* gnn1_bt  = (const float*)d_in[25];
  const float* gnn2_w   = (const float*)d_in[26];
  const float* gnn2_b   = (const float*)d_in[27];
  const float* gnn2_g   = (const float*)d_in[28];
  const float* gnn2_bt  = (const float*)d_in[29];
  const float* gf_w     = (const float*)d_in[30];
  const float* gf_b     = (const float*)d_in[31];
  const float* ee_w1    = (const float*)d_in[32];
  const float* ee_b1    = (const float*)d_in[33];
  const float* ee_ln_g  = (const float*)d_in[34];
  const float* ee_ln_b  = (const float*)d_in[35];
  const float* ee_w2    = (const float*)d_in[36];
  const float* ee_b2    = (const float*)d_in[37];
  const float* ee_w3    = (const float*)d_in[38];
  const float* ee_b3    = (const float*)d_in[39];

  const int BN = BATCH * N_NODES;       // 2864
  const size_t N2 = (size_t)N_NODES * N_NODES;

  float* out_final  = (float*)d_out;                    // (8,358,358)
  float* out_static = out_final + (size_t)BATCH * N2;   // (4,358,358)
  float* out_dyn    = out_static + (size_t)G_HEADS * N2;// (8,4,358,358)

  // workspace layout (wq_t first: keeps 32B alignment for v16bf loads)
  float* ws = (float*)d_ws;
  bf16*  wq_t      = (bf16*)ws;              ws += 18432;  // 3*4*32*96 bf16
  float* bq_pad    = ws;                     ws += 384;
  float* womg      = ws;                     ws += 128;
  float* bomg      = ws;                     ws += 8;
  float* node_repr = ws;                     ws += (size_t)BN * 96;
  float* pmean     = ws;                     ws += (size_t)BN * 96;
  float* de0       = ws;                     ws += (size_t)BN * 64;
  float* de1       = ws;                     ws += (size_t)BN * 64;
  float* de2       = ws;                     ws += (size_t)BN * 64;
  float* fwb       = ws;                     ws += (size_t)BN * 4;

  prep_kernel<<<(3 * 4 * 32 * 96 + 255) / 256, 256, 0, stream>>>(
      in_w, in_b, out_w, out_b, wq_t, bq_pad, womg, bomg);
  static_kernel<<<G_HEADS * N_NODES, 128, 0, stream>>>(le1, le2, ge1, ge2, temp, out_static);
  mha_kernel<<<BN, 256, 0, stream>>>(patch, pos_enc, wq_t, bq_pad, womg, bomg,
                                     out_w, out_b, tn_g, tn_b, node_repr, pmean);
  dyn_enc_kernel<<<BN, 128, 0, stream>>>(node_repr, de_w1, de_b1, de_ln1_g, de_ln1_b,
                                         de_w2, de_b2, de_ln2_g, de_ln2_b, de0);
  gnn_kernel<<<BN, 256, 0, stream>>>(de0, gnn1_w, gnn1_b, gnn1_g, gnn1_bt, de1);
  gnn_kernel<<<BN, 256, 0, stream>>>(de1, gnn2_w, gnn2_b, gnn2_g, gnn2_bt, de2);
  dyn_adj_kernel<<<BATCH * G_HEADS * N_NODES, 256, 0, stream>>>(de2, se2, temp, out_dyn);
  fw_kernel<<<(BN * 4 + 255) / 256, 256, 0, stream>>>(pmean, gf_w, gf_b, fwb);
  fuse_edge_kernel<<<(int)((BATCH * N2 + 255) / 256), 256, 0, stream>>>(
      out_static, out_dyn, fwb, ee_w1, ee_b1, ee_ln_g, ee_ln_b, ee_w2, ee_b2, ee_w3, ee_b3,
      out_final);
}
